// NeRF_45277545234712
// MI455X (gfx1250) — matrix operand order repaired and checked by reading, not compiled, not run on hardware
//
#include <hip/hip_runtime.h>

// ---------------------------------------------------------------------------
// NeRF fused MLP for MI455X (gfx1250, wave32, WMMA f16 16x16x32, f32 accum)
// Block = 256 threads (8 waves) = 128 points. Waves partition output columns.
// Activations live in LDS (row-major, padded stride). Weights pre-packed to
// f16 B-fragment order in d_ws by a pack kernel; loaded straight from L2.
// ---------------------------------------------------------------------------

typedef __attribute__((ext_vector_type(16))) _Float16 v16h;
typedef __attribute__((ext_vector_type(8)))  float    v8f;
typedef unsigned int u32;
typedef __attribute__((ext_vector_type(4)))  u32      v4u;

union FragU { v4u q[2]; v16h h; };

#define AS     328      // activation row stride in halves (320 + 8 pad, bank-friendly)
#define MROWS  128      // points per block
#define NTHR   256      // 8 wave32

// Packed-weight offsets in d_ws (units: halves). Layout per layer:
// [(kc, ntile)][lane 0..31][16 halves], lane<16 -> K=kc*32+{0..7,16..23} col=nt*16+lane,
// lane>=16 -> K=kc*32+{8..15,24..31} col=nt*16+lane-16. Matches 16-bit B-fragment layout.
#define OFS_S0   0u        // 64x256
#define OFS_S1   16384u    // 256x256
#define OFS_S2   81920u
#define OFS_S3   147456u
#define OFS_S4   212992u
#define OFS_S5   278528u   // 320x256 (skip layer: [xe(63)|pad|y(256)])
#define OFS_S6   360448u
#define OFS_S7   425984u
#define OFS_R1   491520u   // 256x256
#define OFS_R2   557056u   // 288x128 ([yr1(256)|de(27)|pad])
#define OFS_R3   593920u   // 128x16  (3 valid cols)

struct Params {
    const float* x; const float* d;
    const float* Wd; const float* bd;
    const float* br1; const float* br2; const float* br3;
    const float* bs[8];
    const _Float16* ws;
    float* out;
    int n;
};

// --- weight pack: fp32 [fanin, fanout] row-major -> f16 B-fragment order ----
// dest element idx -> (kc, nt, laneIdx, e); source row remap handles padding
// and the layer-5 concat offset: ks = k<frontValid ? k : (k>=backStart ? k-backShift : -1)
__global__ void nerf_pack_w(const float* __restrict__ W, _Float16* __restrict__ dst,
                            int Kp, int Nout, int srcN,
                            int frontValid, int backStart, int backShift)
{
    int idx = blockIdx.x * 256 + threadIdx.x;
    int total = Kp * Nout;
    if (idx >= total) return;
    int e  = idx & 15;
    int li = (idx >> 4) & 31;
    int ntkc = idx >> 9;
    int ntiles = Nout >> 4;
    int nt = ntkc % ntiles;
    int kc = ntkc / ntiles;
    int n = nt * 16 + (li & 15);
    int k = kc * 32 + ((e >> 3) << 4) + ((li >> 4) << 3) + (e & 7);
    float v = 0.f;
    int ks = (k < frontValid) ? k : ((k >= backStart) ? (k - backShift) : -1);
    if (ks >= 0 && n < srcN) v = W[(size_t)ks * srcN + n];
    dst[idx] = (_Float16)v;
}

// --- one dense layer: D[128 x Nout] = relu?(A[128 x Kp] * W + b) ------------
// Wave `wave` owns NT ntiles (NT*16 cols). 8 M-groups of 16 rows.
template<int NT, bool RELU>
__device__ __forceinline__ void dense_layer(
    _Float16* act, int kBase, int outBase, int Kp, int NoutTot,
    const _Float16* __restrict__ wsW, const float* __restrict__ bias,
    int wave, int lane)
{
    const int lhalf = lane >> 4;       // 0 or 1
    const int l15   = lane & 15;
    const int ntilesTot = NoutTot >> 4;
    const v8f vzero = {0.f,0.f,0.f,0.f,0.f,0.f,0.f,0.f};

    v8f acc[8][NT];
#pragma unroll
    for (int mg = 0; mg < 8; ++mg)
#pragma unroll
        for (int j = 0; j < NT; ++j) acc[mg][j] = vzero;

    const int nchunks = Kp >> 5;
    for (int kc = 0; kc < nchunks; ++kc) {
        // B fragments: 2x global_load_b128 per ntile, contiguous per lane
        FragU B[NT];
#pragma unroll
        for (int j = 0; j < NT; ++j) {
            const _Float16* pb =
                wsW + ((size_t)((kc * ntilesTot + wave * NT + j) * 32 + lane) << 4);
            B[j].q[0] = *(const v4u*)(pb);
            B[j].q[1] = *(const v4u*)(pb + 8);
        }
        const int k0 = kBase + (kc << 5) + (lhalf << 3);
#pragma unroll
        for (int mg = 0; mg < 8; ++mg) {
            // A fragment: two contiguous 16B runs from this lane's row
            FragU A;
            const _Float16* ar = act + (mg * 16 + l15) * AS + k0;
            A.q[0] = *(const v4u*)(ar);
            A.q[1] = *(const v4u*)(ar + 16);
#pragma unroll
            for (int j = 0; j < NT; ++j)
                acc[mg][j] = __builtin_amdgcn_wmma_f32_16x16x32_f16(
                    false, A.h, false, B[j].h, (short)0, acc[mg][j], false, false);
        }
    }
    __syncthreads();   // all reads of act done before anyone overwrites
    // bias + (relu) + f16 store back to act (output col n becomes K-row outBase+n)
#pragma unroll
    for (int j = 0; j < NT; ++j) {
        const int n = wave * NT * 16 + j * 16 + l15;
        const float bv = bias[n];
#pragma unroll
        for (int mg = 0; mg < 8; ++mg) {
#pragma unroll
            for (int r = 0; r < 8; ++r) {
                float v = acc[mg][j][r] + bv;
                if (RELU) v = fmaxf(v, 0.f);
                act[(mg * 16 + r + (lhalf << 3)) * AS + outBase + n] = (_Float16)v;
            }
        }
    }
    __syncthreads();
}

__global__ __launch_bounds__(NTHR) void nerf_main(Params p)
{
    extern __shared__ _Float16 act[];   // [MROWS][AS] halves
    const int t = threadIdx.x;
    const int wave = t >> 5, lane = t & 31;
    const int blk = blockIdx.x;

    // --- positional embedding xe -> act[m][0..62], pad col 63 ---
    if (t < MROWS) {
        const int gm = blk * MROWS + t;
        float xv0 = p.x[gm * 3 + 0], xv1 = p.x[gm * 3 + 1], xv2 = p.x[gm * 3 + 2];
        _Float16* row = act + t * AS;
        row[0] = (_Float16)xv0; row[1] = (_Float16)xv1; row[2] = (_Float16)xv2;
        float f = 3.14159265358979323846f;
        for (int l = 0; l < 10; ++l) {
            float s0, c0, s1, c1, s2, c2;
            __sincosf(f * xv0, &s0, &c0);
            __sincosf(f * xv1, &s1, &c1);
            __sincosf(f * xv2, &s2, &c2);
            row[3 + 6 * l + 0] = (_Float16)s0;
            row[3 + 6 * l + 1] = (_Float16)s1;
            row[3 + 6 * l + 2] = (_Float16)s2;
            row[3 + 6 * l + 3] = (_Float16)c0;
            row[3 + 6 * l + 4] = (_Float16)c1;
            row[3 + 6 * l + 5] = (_Float16)c2;
            f *= 2.f;
        }
        row[63] = (_Float16)0.f;
    }
    __syncthreads();

    // --- stem: y lives at K-rows [64..319] ---
    dense_layer<2, true>(act, 0, 64, 64, 256, p.ws + OFS_S0, p.bs[0], wave, lane);
    {
        const u32 ofs[4] = {OFS_S1, OFS_S2, OFS_S3, OFS_S4};
        for (int i = 0; i < 4; ++i)
            dense_layer<2, true>(act, 64, 64, 256, 256, p.ws + ofs[i], p.bs[1 + i], wave, lane);
    }
    dense_layer<2, true>(act, 0, 64, 320, 256, p.ws + OFS_S5, p.bs[5], wave, lane);  // skip concat
    {
        const u32 ofs[2] = {OFS_S6, OFS_S7};
        for (int i = 0; i < 2; ++i)
            dense_layer<2, true>(act, 64, 64, 256, 256, p.ws + ofs[i], p.bs[6 + i], wave, lane);
    }

    // --- sigma = y . Wd + bd (out[3N + gm]) ---
    if (t < MROWS) {
        const int gm = blk * MROWS + t;
        const _Float16* row = act + t * AS + 64;
        float s = p.bd[0];
        for (int k = 0; k < 256; k += 8) {
            union { v4u q; _Float16 h[8]; } u;
            u.q = *(const v4u*)(row + k);
#pragma unroll
            for (int r = 0; r < 8; ++r) s += (float)u.h[r] * p.Wd[k + r];
        }
        p.out[(size_t)3 * p.n + gm] = s;
    }

    // --- Wr1 (no relu): y(rows 64..319) -> yr1 at rows [0..255] ---
    dense_layer<2, false>(act, 64, 0, 256, 256, p.ws + OFS_R1, p.br1, wave, lane);

    // --- direction embedding de -> rows [256..282], pad to 287 ---
    if (t < MROWS) {
        const int gm = blk * MROWS + t;
        float dv0 = p.d[gm * 3 + 0], dv1 = p.d[gm * 3 + 1], dv2 = p.d[gm * 3 + 2];
        _Float16* row = act + t * AS + 256;
        row[0] = (_Float16)dv0; row[1] = (_Float16)dv1; row[2] = (_Float16)dv2;
        float f = 3.14159265358979323846f;
        for (int l = 0; l < 4; ++l) {
            float s0, c0, s1, c1, s2, c2;
            __sincosf(f * dv0, &s0, &c0);
            __sincosf(f * dv1, &s1, &c1);
            __sincosf(f * dv2, &s2, &c2);
            row[3 + 6 * l + 0] = (_Float16)s0;
            row[3 + 6 * l + 1] = (_Float16)s1;
            row[3 + 6 * l + 2] = (_Float16)s2;
            row[3 + 6 * l + 3] = (_Float16)c0;
            row[3 + 6 * l + 4] = (_Float16)c1;
            row[3 + 6 * l + 5] = (_Float16)c2;
            f *= 2.f;
        }
        for (int z = 27; z < 32; ++z) row[z] = (_Float16)0.f;
    }
    __syncthreads();

    // --- Wr2 (relu): [yr1|de] rows 0..287 -> rows [0..127] ---
    dense_layer<1, true>(act, 0, 0, 288, 128, p.ws + OFS_R2, p.br2, wave, lane);

    // --- Wr3: rows 0..127 -> c (3 cols, padded to one 16-col WMMA tile) ---
    {
        const int lhalf = lane >> 4, l15 = lane & 15;
        v8f acc = {0.f,0.f,0.f,0.f,0.f,0.f,0.f,0.f};
        for (int kc = 0; kc < 4; ++kc) {
            FragU B;
            const _Float16* pb = p.ws + OFS_R3 + ((size_t)(kc * 32 + lane) << 4);
            B.q[0] = *(const v4u*)(pb);
            B.q[1] = *(const v4u*)(pb + 8);
            FragU A;
            const _Float16* ar = act + (wave * 16 + l15) * AS + (kc << 5) + (lhalf << 3);
            A.q[0] = *(const v4u*)(ar);
            A.q[1] = *(const v4u*)(ar + 16);
            acc = __builtin_amdgcn_wmma_f32_16x16x32_f16(
                false, A.h, false, B.h, (short)0, acc, false, false);
        }
        const int n = l15;
        if (n < 3) {
            const float bv = p.br3[n];
#pragma unroll
            for (int r = 0; r < 8; ++r) {
                const int m = wave * 16 + r + (lhalf << 3);
                const int gm = blk * MROWS + m;
                p.out[(size_t)gm * 3 + n] = acc[r] + bv;
            }
        }
    }
}

extern "C" void kernel_launch(void* const* d_in, const int* in_sizes, int n_in,
                              void* d_out, int out_size, void* d_ws, size_t ws_size,
                              hipStream_t stream)
{
    // d_in order: x, d, W_stem[0..7], b_stem[0..7], Wd, bd, Wr1, br1, Wr2, br2, Wr3, br3
    const float* x = (const float*)d_in[0];
    const float* d = (const float*)d_in[1];
    const float* Ws[8]; const float* bs[8];
    for (int i = 0; i < 8; ++i) { Ws[i] = (const float*)d_in[2 + i]; bs[i] = (const float*)d_in[10 + i]; }
    const float* Wd  = (const float*)d_in[18];
    const float* bd  = (const float*)d_in[19];
    const float* Wr1 = (const float*)d_in[20];
    const float* br1 = (const float*)d_in[21];
    const float* Wr2 = (const float*)d_in[22];
    const float* br2 = (const float*)d_in[23];
    const float* Wr3 = (const float*)d_in[24];
    const float* br3 = (const float*)d_in[25];
    const int N = in_sizes[0] / 3;

    _Float16* ws = (_Float16*)d_ws;
    const int BIG = 1 << 30;
    auto pack = [&](const float* W, u32 ofs, int Kp, int Nout, int srcN,
                    int fv, int bstart, int bshift) {
        int total = Kp * Nout;
        nerf_pack_w<<<(total + 255) / 256, 256, 0, stream>>>(
            W, ws + ofs, Kp, Nout, srcN, fv, bstart, bshift);
    };
    pack(Ws[0], OFS_S0,  64, 256, 256,  63, BIG, 0);          // 63 real rows + pad
    pack(Ws[1], OFS_S1, 256, 256, 256, 256, BIG, 0);
    pack(Ws[2], OFS_S2, 256, 256, 256, 256, BIG, 0);
    pack(Ws[3], OFS_S3, 256, 256, 256, 256, BIG, 0);
    pack(Ws[4], OFS_S4, 256, 256, 256, 256, BIG, 0);
    pack(Ws[5], OFS_S5, 320, 256, 256,  63,  64, 1);          // [xe|pad|y] remap
    pack(Ws[6], OFS_S6, 256, 256, 256, 256, BIG, 0);
    pack(Ws[7], OFS_S7, 256, 256, 256, 256, BIG, 0);
    pack(Wr1,   OFS_R1, 256, 256, 256, 256, BIG, 0);
    pack(Wr2,   OFS_R2, 288, 128, 128, 283, BIG, 0);          // 283 real rows + pad
    pack(Wr3,   OFS_R3, 128,  16,   3, 128, BIG, 0);          // 3 real cols, pad to 16

    Params p;
    p.x = x; p.d = d; p.Wd = Wd; p.bd = bd;
    p.br1 = br1; p.br2 = br2; p.br3 = br3;
    for (int i = 0; i < 8; ++i) p.bs[i] = bs[i];
    p.ws = ws; p.out = (float*)d_out; p.n = N;

    const size_t smem = (size_t)MROWS * AS * sizeof(_Float16);  // 83968 B (<320KB WGP LDS)
    (void)hipFuncSetAttribute((const void*)nerf_main,
                              hipFuncAttributeMaxDynamicSharedMemorySize, (int)smem);
    nerf_main<<<N / MROWS, NTHR, smem, stream>>>(p);
}
